// MuActor_69690139345310
// MI455X (gfx1250) — matrix operand order repaired and checked by reading, not compiled, not run on hardware
//
#include <hip/hip_runtime.h>
#include <hip/hip_bf16.h>

typedef __attribute__((ext_vector_type(16))) _Float16 v16h;
typedef __attribute__((ext_vector_type(8)))  _Float16 v8h;
typedef __attribute__((ext_vector_type(8)))  float    v8f;

namespace {
constexpr int S_DIM  = 376;
constexpr int H1_DIM = 400;
constexpr int H2_DIM = 300;
constexpr int A_DIM  = 17;
constexpr int K_FLOW = 15;

// padded GEMM dims
constexpr int KP0 = 384;   // S padded to mult of 32
constexpr int NP1 = 400;   // H1 (already mult of 16)
constexpr int KP1 = 416;   // H1 padded to mult of 32
constexpr int NP2 = 304;   // H2 padded to mult of 16
constexpr int KP2 = 320;   // H2 padded to mult of 32
constexpr int NPH = 560;   // 559 head cols padded to mult of 16

// LDS row strides (elements), padded so A-fragment ds_load_b128 are conflict-free
constexpr int ST0 = KP0 + 8;   // f16
constexpr int ST1 = KP1 + 8;   // f16
constexpr int ST2 = KP2 + 8;   // f16
constexpr int PST = NPH + 4;   // f32

constexpr int XS_TILE  = 16 * ST0 * 2;   // 12544 B
constexpr int H1_TILE  = 16 * ST1 * 2;   // 13568 B
constexpr int H2_TILE  = 16 * ST2 * 2;   // 10496 B
constexpr int PH_BYTES = 16 * PST * 4;   // 36096 B

// LDS overlay schedule (all offsets 16B aligned):
//   stage+L1 : xs[2] @ 0        (25088) | h1s[2] @ 25088 (27136)
//   L2       : h2s[2] @ 0       (20992, over dead xs) | h1s[2] @ 25088
//   heads    : h2s[2] @ 0       | ph @ 20992 (36096, over dead h1s)
constexpr int OFF_XS = 0;
constexpr int OFF_H2 = 0;
constexpr int OFF_H1 = 2 * XS_TILE;           // 25088
constexpr int OFF_PH = 2 * H2_TILE;           // 20992
constexpr int SMEM_BYTES = OFF_PH + PH_BYTES; // 57088
static_assert(OFF_H1 + 2 * H1_TILE <= SMEM_BYTES, "h1 fits");
static_assert(SMEM_BYTES <= 64 * 1024, "LDS budget");

constexpr size_t align256(size_t x) { return (x + 255) / 256 * 256; }
constexpr size_t WS_W1T = 0;
constexpr size_t WS_W2T = WS_W1T + align256((size_t)NP1 * KP0 * 2);
constexpr size_t WS_WHT = WS_W2T + align256((size_t)NP2 * KP1 * 2);
constexpr size_t WS_BH  = WS_WHT + align256((size_t)NPH * KP2 * 2);
} // namespace

// ---------------------------------------------------------------------------
// Prep: transpose weights to (N x Kpad) f16 with zero padding; concat head bias
// ---------------------------------------------------------------------------
__global__ void prep_kernel(const float* __restrict__ W1, const float* __restrict__ W2,
                            const float* __restrict__ Wmu, const float* __restrict__ Wlv,
                            const float* __restrict__ Wu,  const float* __restrict__ Ww,
                            const float* __restrict__ Wb,
                            const float* __restrict__ bmu, const float* __restrict__ blv,
                            const float* __restrict__ bu,  const float* __restrict__ bw,
                            const float* __restrict__ bb,
                            _Float16* __restrict__ W1t, _Float16* __restrict__ W2t,
                            _Float16* __restrict__ Wht, float* __restrict__ bh)
{
    const int total = NP1 * KP0 + NP2 * KP1 + NPH * KP2 + NPH;
    for (int i = blockIdx.x * blockDim.x + threadIdx.x; i < total;
         i += gridDim.x * blockDim.x) {
        int j = i;
        if (j < NP1 * KP0) {
            int n = j / KP0, k = j % KP0;
            W1t[j] = (k < S_DIM) ? (_Float16)W1[k * H1_DIM + n] : (_Float16)0.f;
            continue;
        }
        j -= NP1 * KP0;
        if (j < NP2 * KP1) {
            int n = j / KP1, k = j % KP1;
            W2t[j] = (n < H2_DIM && k < H1_DIM) ? (_Float16)W2[k * H2_DIM + n]
                                                : (_Float16)0.f;
            continue;
        }
        j -= NP2 * KP1;
        if (j < NPH * KP2) {
            int n = j / KP2, k = j % KP2;
            float v = 0.f;
            if (k < H2_DIM) {
                if (n < 17)       v = Wmu[k * 17 + n];
                else if (n < 34)  v = Wlv[k * 17 + (n - 17)];
                else if (n < 289) v = Wu[k * (K_FLOW * A_DIM) + (n - 34)];
                else if (n < 544) v = Ww[k * (K_FLOW * A_DIM) + (n - 289)];
                else if (n < 559) v = Wb[k * K_FLOW + (n - 544)];
            }
            Wht[j] = (_Float16)v;
            continue;
        }
        j -= NPH * KP2;
        {
            int n = j;
            float v = 0.f;
            if (n < 17)       v = bmu[n];
            else if (n < 34)  v = blv[n - 17];
            else if (n < 289) v = bu[n - 34];
            else if (n < 544) v = bw[n - 289];
            else if (n < 559) v = bb[n - 544];
            bh[n] = v;
        }
    }
}

// A-fragment layout: lanes {0-15,16-31} hold M=lane&15; VGPR0-3 = K
// [hi*16, hi*16+8), VGPR4-7 = K [hi*16+8, hi*16+16).
union AFrag { v16h v; v8h h[2]; };

__device__ __forceinline__ v16h load_a_lds(const _Float16* base, int stride,
                                           int m, int kbase, int hi)
{
    AFrag a;
    a.h[0] = *(const v8h*)(base + m * stride + kbase + hi * 16);
    a.h[1] = *(const v8h*)(base + m * stride + kbase + hi * 16 + 8);
    return a.v;
}

__device__ __forceinline__ v8h cvt8(float4 a, float4 b)
{
    v8h r;
    r[0] = (_Float16)a.x; r[1] = (_Float16)a.y; r[2] = (_Float16)a.z; r[3] = (_Float16)a.w;
    r[4] = (_Float16)b.x; r[5] = (_Float16)b.y; r[6] = (_Float16)b.z; r[7] = (_Float16)b.w;
    return r;
}

// ---------------------------------------------------------------------------
// Fused actor forward: 1 wave per 32-row block (two 16-row M-tiles).
// Each B fragment read from L2 feeds two WMMAs (halves weight traffic).
// ---------------------------------------------------------------------------
__global__ void __launch_bounds__(32)
actor_kernel(const float* __restrict__ x, const float* __restrict__ eps,
             const float* __restrict__ b1, const float* __restrict__ b2,
             const _Float16* __restrict__ W1t, const _Float16* __restrict__ W2t,
             const _Float16* __restrict__ Wht, const float* __restrict__ bh,
             float* __restrict__ out, int Bn)
{
    __shared__ __align__(16) unsigned char smem[SMEM_BYTES];
    _Float16* xs  = (_Float16*)(smem + OFF_XS);   // 32 x ST0; dead after layer1
    _Float16* h1s = (_Float16*)(smem + OFF_H1);   // 32 x ST1; dead after layer2
    _Float16* h2s = (_Float16*)(smem + OFF_H2);   // 32 x ST2; overlays dead xs
    float*    ph  = (float*)(smem + OFF_PH);      // 16 x PST; overlays dead h1s

    const int lane = threadIdx.x;
    const int ln = lane & 15, hi = lane >> 4;
    const long row0 = (long)blockIdx.x * 32;

    // ---- stage x (32 rows) -> LDS f16, zero-pad K to 384 ----
    for (int r = 0; r < 32; ++r) {
        const float* xr = x + (row0 + r) * (long)S_DIM;
        _Float16* xd = xs + r * ST0;
        for (int c8 = lane; c8 < S_DIM / 8; c8 += 32) {       // 376 = 47*8
            float4 v0 = ((const float4*)xr)[c8 * 2];
            float4 v1 = ((const float4*)xr)[c8 * 2 + 1];
            *(v8h*)(xd + c8 * 8) = cvt8(v0, v1);
        }
    }
    for (int i = lane; i < 32 * (KP0 - S_DIM); i += 32) {     // zero K-pad 376..383
        int r = i / (KP0 - S_DIM), c = S_DIM + i % (KP0 - S_DIM);
        xs[r * ST0 + c] = (_Float16)0.f;
    }
    __syncthreads();

    // ---- layer 1: h1 = relu(x @ W1 + b1), 25 N-tiles x 12 K-steps x 2 M ----
    for (int nt = 0; nt < NP1 / 16; ++nt) {
        v8f acc0 = {}, acc1 = {};
        const int n = nt * 16 + ln;
        const _Float16* wrow = W1t + (size_t)n * KP0 + hi * 16;
        for (int kt = 0; kt < KP0 / 32; ++kt) {
            v16h b  = *(const v16h*)(wrow + kt * 32);
            v16h a0 = load_a_lds(xs, ST0, ln,      kt * 32, hi);
            v16h a1 = load_a_lds(xs, ST0, 16 + ln, kt * 32, hi);
            acc0 = __builtin_amdgcn_wmma_f32_16x16x32_f16(false, a0, false, b,
                                                          (short)0, acc0, false, false);
            acc1 = __builtin_amdgcn_wmma_f32_16x16x32_f16(false, a1, false, b,
                                                          (short)0, acc1, false, false);
        }
        const float bias = b1[n];
#pragma unroll
        for (int r = 0; r < 8; ++r) {
            const int m = hi * 8 + r;
            h1s[m * ST1 + n]        = (_Float16)fmaxf(acc0[r] + bias, 0.f);
            h1s[(16 + m) * ST1 + n] = (_Float16)fmaxf(acc1[r] + bias, 0.f);
        }
    }
    for (int i = lane; i < 32 * (KP1 - H1_DIM); i += 32) {    // zero K-pad 400..415
        int r = i / (KP1 - H1_DIM), c = H1_DIM + i % (KP1 - H1_DIM);
        h1s[r * ST1 + c] = (_Float16)0.f;
    }
    __syncthreads();   // xs dead from here; h2s may overwrite it

    // ---- layer 2: h2 = relu(h1 @ W2 + b2), 19 N-tiles x 13 K-steps x 2 M ----
    for (int nt = 0; nt < NP2 / 16; ++nt) {
        v8f acc0 = {}, acc1 = {};
        const int n = nt * 16 + ln;
        const _Float16* wrow = W2t + (size_t)n * KP1 + hi * 16;
        for (int kt = 0; kt < KP1 / 32; ++kt) {
            v16h b  = *(const v16h*)(wrow + kt * 32);
            v16h a0 = load_a_lds(h1s, ST1, ln,      kt * 32, hi);
            v16h a1 = load_a_lds(h1s, ST1, 16 + ln, kt * 32, hi);
            acc0 = __builtin_amdgcn_wmma_f32_16x16x32_f16(false, a0, false, b,
                                                          (short)0, acc0, false, false);
            acc1 = __builtin_amdgcn_wmma_f32_16x16x32_f16(false, a1, false, b,
                                                          (short)0, acc1, false, false);
        }
        const int nc = (n < H2_DIM) ? n : 0;
        float bias = b2[nc];
        if (n >= H2_DIM) bias = 0.f;
#pragma unroll
        for (int r = 0; r < 8; ++r) {
            const int m = hi * 8 + r;
            float v0 = fmaxf(acc0[r] + bias, 0.f);
            float v1 = fmaxf(acc1[r] + bias, 0.f);
            h2s[m * ST2 + n]        = (n < H2_DIM) ? (_Float16)v0 : (_Float16)0.f;
            h2s[(16 + m) * ST2 + n] = (n < H2_DIM) ? (_Float16)v1 : (_Float16)0.f;
        }
    }
    for (int i = lane; i < 32 * (KP2 - NP2); i += 32) {       // zero K-pad 304..319
        int r = i / (KP2 - NP2), c = NP2 + i % (KP2 - NP2);
        h2s[r * ST2 + c] = (_Float16)0.f;
    }

    // ---- heads + planar-flow epilogue, one 16-row half at a time ----
    for (int t = 0; t < 2; ++t) {
        __syncthreads();   // h1s dead (t=0) / previous epilogue done (t=1)
        const _Float16* h2t = h2s + t * 16 * ST2;

        // ph = h2 @ [Wmu|Wlv|Wu|Ww|Wb] + bias, 35 N-tiles x 10 K-steps
        for (int nt = 0; nt < NPH / 16; ++nt) {
            v8f acc = {};
            const int n = nt * 16 + ln;
            const _Float16* wrow = Wht + (size_t)n * KP2 + hi * 16;
            for (int kt = 0; kt < KP2 / 32; ++kt) {
                v16h b = *(const v16h*)(wrow + kt * 32);
                v16h a = load_a_lds(h2t, ST2, ln, kt * 32, hi);
                acc = __builtin_amdgcn_wmma_f32_16x16x32_f16(false, a, false, b,
                                                             (short)0, acc, false, false);
            }
            const float bias = bh[n];
#pragma unroll
            for (int r = 0; r < 8; ++r) {
                const int m = hi * 8 + r;
                ph[m * PST + n] = acc[r] + bias;
            }
        }
        __syncthreads();

        // flow chain + log-prob: one row per lane 0..15
        if (lane < 16) {
            const long row = row0 + t * 16 + lane;
            const float* phr = ph + lane * PST;
            const float* er  = eps + row * A_DIM;

            float z[A_DIM];
            float lp = 0.f;
#pragma unroll
            for (int a = 0; a < A_DIM; ++a) {
                const float e   = er[a];
                const float mu  = tanhf(phr[a]);          // MAX_ACTION = 1.0
                const float lvv = tanhf(phr[17 + a]);
                z[a] = mu + expf(lvv) * e;
                lp += -0.5f * e * e - lvv - 0.91893853320467274f; // 0.5*log(2*pi)
            }

            float ldj = 0.f;
            for (int k = 0; k < K_FLOW; ++k) {
                const float* u = phr + 34 + k * A_DIM;
                const float* w = phr + 289 + k * A_DIM;
                const float bk = phr[544 + k];
                float uw = 0.f, wns = 0.f, wz = 0.f;
#pragma unroll
                for (int a = 0; a < A_DIM; ++a) {
                    const float wa = w[a];
                    uw  += wa * u[a];
                    wns += wa * wa;
                    wz  += wa * z[a];
                }
                const float sp   = fmaxf(uw, 0.f) + log1pf(expf(-fabsf(uw)));
                const float coef = (-1.f + sp - uw) / wns;
                const float t2   = tanhf(wz + bk);
#pragma unroll
                for (int a = 0; a < A_DIM; ++a)
                    z[a] += (u[a] + coef * w[a]) * t2;
                const float wuhat = uw + coef * wns;      // w . u_hat
                ldj += logf(fabsf(1.f + wuhat * (1.f - t2 * t2)));
            }

            float* oa = out + row * A_DIM;
#pragma unroll
            for (int a = 0; a < A_DIM; ++a) oa[a] = z[a];
            const float lpf = lp - ldj;
            out[(long)Bn * A_DIM + row]       = expf(lpf);   // prob_final
            out[(long)Bn * (A_DIM + 1) + row] = lpf;         // log_prob_final
        }
    }
}

extern "C" void kernel_launch(void* const* d_in, const int* in_sizes, int n_in,
                              void* d_out, int out_size, void* d_ws, size_t ws_size,
                              hipStream_t stream)
{
    (void)n_in; (void)out_size; (void)ws_size;
    const float* x   = (const float*)d_in[0];
    const float* eps = (const float*)d_in[1];
    const float* W1  = (const float*)d_in[2];
    const float* b1  = (const float*)d_in[3];
    const float* W2  = (const float*)d_in[4];
    const float* b2  = (const float*)d_in[5];
    const float* Wmu = (const float*)d_in[6];
    const float* bmu = (const float*)d_in[7];
    const float* Wlv = (const float*)d_in[8];
    const float* blv = (const float*)d_in[9];
    const float* Wu  = (const float*)d_in[10];
    const float* bu  = (const float*)d_in[11];
    const float* Ww  = (const float*)d_in[12];
    const float* bw  = (const float*)d_in[13];
    const float* Wb  = (const float*)d_in[14];
    const float* bb  = (const float*)d_in[15];

    const int Bn = in_sizes[0] / S_DIM;   // 65536

    char* ws = (char*)d_ws;
    _Float16* W1t = (_Float16*)(ws + WS_W1T);
    _Float16* W2t = (_Float16*)(ws + WS_W2T);
    _Float16* Wht = (_Float16*)(ws + WS_WHT);
    float*    bh  = (float*)(ws + WS_BH);

    prep_kernel<<<1024, 256, 0, stream>>>(W1, W2, Wmu, Wlv, Wu, Ww, Wb,
                                          bmu, blv, bu, bw, bb,
                                          W1t, W2t, Wht, bh);
    actor_kernel<<<Bn / 32, 32, 0, stream>>>(x, eps, b1, b2, W1t, W2t, Wht, bh,
                                             (float*)d_out, Bn);
}
